// ScopeRecognizer_31988916420769
// MI455X (gfx1250) — matrix operand order repaired and checked by reading, not compiled, not run on hardware
//
#include <hip/hip_runtime.h>
#include <math.h>

// ---------------------------------------------------------------------------
// Problem constants (from reference): B=4, L=2048, H=1024, band |i-j| <= 4
// ---------------------------------------------------------------------------
constexpr int kB    = 4;
constexpr int kL    = 2048;
constexpr int kH    = 1024;
constexpr int kBand = 4;          // |i-j| <= 4  -> 9 diagonals

// GEMM tiling
constexpr int BM  = 64;           // rows per block
constexpr int BN  = 128;          // cols per block
constexpr int BKC = 32;           // K chunk (one bf16 WMMA K)

typedef __attribute__((ext_vector_type(16))) __bf16 v16bf;
typedef __attribute__((ext_vector_type(8)))  float  v8f;
typedef __attribute__((ext_vector_type(4)))  int    v4i;

union Frag {
    v16bf v;
    unsigned int u[8];
};

__device__ __forceinline__ unsigned short bf16_rn(float f) {
    unsigned int u = __float_as_uint(f);
    unsigned int r = u + 0x7FFFu + ((u >> 16) & 1u);   // round-to-nearest-even
    return (unsigned short)(r >> 16);
}
__device__ __forceinline__ float bf16_to_f32(unsigned short h) {
    return __uint_as_float(((unsigned int)h) << 16);
}

// ---------------------------------------------------------------------------
// Async global->LDS copy (CDNA5): GLOBAL_LOAD_ASYNC_TO_LDS_B128 via builtin,
// tracked with ASYNCcnt.  Builtin signature (probe-confirmed by the round-2
// diagnostic): (int4 addrspace(1)* src, int4 addrspace(3)* dst, imm offset,
// imm cpol).  Guarded fallback keeps the file compiling on other toolchains.
// ---------------------------------------------------------------------------
#if defined(__HIP_DEVICE_COMPILE__) && __has_builtin(__builtin_amdgcn_global_load_async_to_lds_b128)
#define HAVE_ASYNC 1
#else
#define HAVE_ASYNC 0
#endif

typedef __attribute__((address_space(1))) void gas_void;
typedef __attribute__((address_space(3))) void las_void;
typedef __attribute__((address_space(1))) v4i* gv4i_p;
typedef __attribute__((address_space(3))) v4i* lv4i_p;

__device__ __forceinline__ void copy16(const unsigned short* g, unsigned short* l) {
#if HAVE_ASYNC
    __builtin_amdgcn_global_load_async_to_lds_b128(
        (gv4i_p)(gas_void*)g, (lv4i_p)(las_void*)l, 0, 0);
#else
    *(uint4*)l = *(const uint4*)g;
#endif
}

__device__ __forceinline__ void wait_async_le6() {
#if HAVE_ASYNC
#if __has_builtin(__builtin_amdgcn_s_wait_asynccnt)
    __builtin_amdgcn_s_wait_asynccnt(6);
#else
    asm volatile("s_wait_asynccnt 0x6" ::: "memory");
#endif
#endif
}
__device__ __forceinline__ void wait_async_0() {
#if HAVE_ASYNC
#if __has_builtin(__builtin_amdgcn_s_wait_asynccnt)
    __builtin_amdgcn_s_wait_asynccnt(0);
#else
    asm volatile("s_wait_asynccnt 0x0" ::: "memory");
#endif
#endif
}

// ---------------------------------------------------------------------------
// Kernel 0: one-shot split of f32 tensors into bf16 hi/lo planes so the GEMM
// hot loop has zero conversion VALU.  x = hi + lo gives ~16 mantissa bits.
// ---------------------------------------------------------------------------
__global__ __launch_bounds__(256)
void split_convert_kernel(const float* __restrict__ src,
                          unsigned short* __restrict__ hi,
                          unsigned short* __restrict__ lo, int n4) {
    const int idx = blockIdx.x * 256 + threadIdx.x;
    if (idx >= n4) return;
    float4 f = ((const float4*)src)[idx];
    ushort4 h, l;
    h.x = bf16_rn(f.x); l.x = bf16_rn(f.x - bf16_to_f32(h.x));
    h.y = bf16_rn(f.y); l.y = bf16_rn(f.y - bf16_to_f32(h.y));
    h.z = bf16_rn(f.z); l.z = bf16_rn(f.z - bf16_to_f32(h.z));
    h.w = bf16_rn(f.w); l.w = bf16_rn(f.w - bf16_to_f32(h.w));
    ((ushort4*)hi)[idx] = h;
    ((ushort4*)lo)[idx] = l;
}

// ---------------------------------------------------------------------------
// Kernel 1: Y = X @ W^T + bias via split-bf16 WMMA.
// A/B tiles are streamed into LDS with double-buffered async copies
// (6 x B128 per thread per stage = 24 KB/stage); inner loop is
// ds_load_b128 fragments + 3x v_wmma_f32_16x16x32_bf16 per tile.
// blockIdx.z: 0 -> (q,Wq,bq)->qs ; 1 -> (k,Wk,bk)->ks
// ---------------------------------------------------------------------------
__global__ __launch_bounds__(256)
void proj_gemm_kernel(const unsigned short* __restrict__ qhi, const unsigned short* __restrict__ qlo,
                      const unsigned short* __restrict__ khi, const unsigned short* __restrict__ klo,
                      const unsigned short* __restrict__ Wqhi, const unsigned short* __restrict__ Wqlo,
                      const unsigned short* __restrict__ Wkhi, const unsigned short* __restrict__ Wklo,
                      const float* __restrict__ bq, const float* __restrict__ bk,
                      float* __restrict__ qs, float* __restrict__ ks) {
    const unsigned short *Xhi, *Xlo, *Whi, *Wlo; const float* bias; float* Y;
    if (blockIdx.z == 0) { Xhi = qhi; Xlo = qlo; Whi = Wqhi; Wlo = Wqlo; bias = bq; Y = qs; }
    else                 { Xhi = khi; Xlo = klo; Whi = Wkhi; Wlo = Wklo; bias = bk; Y = ks; }

    __shared__ __align__(16) unsigned short sAhi[2][BM][BKC];
    __shared__ __align__(16) unsigned short sAlo[2][BM][BKC];
    __shared__ __align__(16) unsigned short sBhi[2][BN][BKC];
    __shared__ __align__(16) unsigned short sBlo[2][BN][BKC];

    const int tid  = threadIdx.x;
    const int wave = tid >> 5;               // 0..7
    const int lane = tid & 31;

    const int m_block = blockIdx.x * BM;     // grid.x = 8192/64 = 128
    const int n_block = blockIdx.y * BN;     // grid.y = 1024/128 = 8
    const int m_off   = (wave & 3) * 16;
    const int n_off   = (wave >> 2) * 64;

    // --- per-thread copy assignment: 16B segments of 64B rows -------------
    const int ra = tid >> 2;                 // A row 0..63 (also B rows ra, ra+64)
    const int sg = (tid & 3) * 8;            // segment offset in elements

    const unsigned short* gAhi = Xhi + (size_t)(m_block + ra) * kH + sg;
    const unsigned short* gAlo = Xlo + (size_t)(m_block + ra) * kH + sg;
    const unsigned short* gBh0 = Whi + (size_t)(n_block + ra) * kH + sg;
    const unsigned short* gBh1 = gBh0 + (size_t)64 * kH;
    const unsigned short* gBl0 = Wlo + (size_t)(n_block + ra) * kH + sg;
    const unsigned short* gBl1 = gBl0 + (size_t)64 * kH;

    v8f acc[4] = {};

    auto stage = [&](int buf, int k0) {
        copy16(gAhi + k0, &sAhi[buf][ra][sg]);
        copy16(gAlo + k0, &sAlo[buf][ra][sg]);
        copy16(gBh0 + k0, &sBhi[buf][ra][sg]);
        copy16(gBh1 + k0, &sBhi[buf][ra + 64][sg]);
        copy16(gBl0 + k0, &sBlo[buf][ra][sg]);
        copy16(gBl1 + k0, &sBlo[buf][ra + 64][sg]);
    };

    constexpr int NCH = kH / BKC;            // 32 K-chunks
    stage(0, 0);
    int buf = 0;

    for (int it = 0; it < NCH; ++it) {
        if (it + 1 < NCH) {                  // prefetch next chunk, then retire current
            stage(buf ^ 1, (it + 1) * BKC);
            wait_async_le6();                // older 6 copies (chunk `it`) complete
        } else {
            wait_async_0();
        }
        __syncthreads();

        // A fragment (16x32 bf16, CDNA5 wave32 layout):
        // lanes 0-15: M=lane,    K in {0..7,16..23}; lanes 16-31: M=lane-16, K in {8..15,24..31}
        Frag ahi, alo;
        {
            const int mloc = m_off + (lane & 15);
            const int kb   = (lane & 16) ? 8 : 0;
            #pragma unroll
            for (int v = 0; v < 8; ++v) {
                const int kk = kb + ((v >> 2) << 4) + ((v & 3) << 1);
                ahi.u[v] = *(const unsigned int*)&sAhi[buf][mloc][kk];
                alo.u[v] = *(const unsigned int*)&sAlo[buf][mloc][kk];
            }
        }
        // B 32x16 layout: lanes 0-15: N=lane, K in {0..7,16..23} ; lanes 16-31: N=lane-16, K in {8..15,24..31}
        #pragma unroll
        for (int t = 0; t < 4; ++t) {
            Frag bhi, blo;
            const int nloc = n_off + t * 16 + (lane & 15);
            const int kb2  = (lane & 16) ? 8 : 0;
            #pragma unroll
            for (int v = 0; v < 8; ++v) {
                const int kk = kb2 + ((v >> 2) << 4) + ((v & 3) << 1);
                bhi.u[v] = *(const unsigned int*)&sBhi[buf][nloc][kk];
                blo.u[v] = *(const unsigned int*)&sBlo[buf][nloc][kk];
            }
            acc[t] = __builtin_amdgcn_wmma_f32_16x16x32_bf16(
                false, alo.v, false, bhi.v, (short)0, acc[t], false, false);
            acc[t] = __builtin_amdgcn_wmma_f32_16x16x32_bf16(
                false, ahi.v, false, blo.v, (short)0, acc[t], false, false);
            acc[t] = __builtin_amdgcn_wmma_f32_16x16x32_bf16(
                false, ahi.v, false, bhi.v, (short)0, acc[t], false, false);
        }
        __syncthreads();                     // everyone done reading buf before it is refilled
        buf ^= 1;
    }

    // epilogue: add per-column bias, store f32
    // C/D layout: VGPR v, lanes 0-15 -> M=v, N=lane ; lanes 16-31 -> M=v+8
    const int mrow = (lane & 16) ? 8 : 0;
    #pragma unroll
    for (int t = 0; t < 4; ++t) {
        const int ng = n_block + n_off + t * 16 + (lane & 15);
        const float bv = bias[ng];
        #pragma unroll
        for (int v = 0; v < 8; ++v) {
            const int mg = m_block + m_off + mrow + v;
            Y[(size_t)mg * kH + ng] = acc[t][v] + bv;
        }
    }
}

// ---------------------------------------------------------------------------
// Kernel 2: banded logits + softmax + dense store.
// One 288-thread block (9 waves) per output row; wave w computes the dot for
// j=i-4+w, wave32 shuffle-reduce, 9-way softmax, then one coalesced
// non-temporal pass writes the 2048-float row (zeros + band probabilities).
// ---------------------------------------------------------------------------
__global__ __launch_bounds__(288)
void band_softmax_kernel(const float* __restrict__ qs, const float* __restrict__ ks,
                         float* __restrict__ out) {
    const int ri   = blockIdx.x;           // b*L + i
    const int b    = ri >> 11;
    const int i    = ri & (kL - 1);
    const int tid  = threadIdx.x;
    const int wave = tid >> 5;             // 0..8
    const int lane = tid & 31;

    __shared__ float s_logit[9];
    __shared__ float s_prob[9];

    const int j = i - kBand + wave;
    const bool valid = (j >= 0) && (j < kL);
    float sum = 0.0f;
    if (valid) {
        const float* qr = qs + (size_t)ri * kH;
        const float* kr = ks + (size_t)(b * kL + j) * kH;
        #pragma unroll
        for (int it = 0; it < kH / 128; ++it) {
            const int h = it * 128 + lane * 4;
            float4 a = *(const float4*)(qr + h);
            float4 c = *(const float4*)(kr + h);
            sum += a.x * c.x + a.y * c.y + a.z * c.z + a.w * c.w;
        }
    }
    #pragma unroll
    for (int off = 16; off > 0; off >>= 1)
        sum += __shfl_xor(sum, off, 32);
    if (lane == 0) s_logit[wave] = valid ? sum : -INFINITY;
    __syncthreads();

    if (tid == 0) {
        float m = -INFINITY;
        #pragma unroll
        for (int w = 0; w < 9; ++w) m = fmaxf(m, s_logit[w]);
        float e[9];
        float tot = 0.0f;
        #pragma unroll
        for (int w = 0; w < 9; ++w) { e[w] = __expf(s_logit[w] - m); tot += e[w]; }
        const float inv = 1.0f / tot;
        #pragma unroll
        for (int w = 0; w < 9; ++w) s_prob[w] = e[w] * inv;
    }
    __syncthreads();

    float* orow = out + (size_t)ri * kL;
    for (int c = tid; c < kL; c += 288) {
        const int d = c - i + kBand;                 // 0..8 iff in band
        const float vout = ((unsigned)d < 9u) ? s_prob[d] : 0.0f;
        __builtin_nontemporal_store(vout, &orow[c]); // 67 MB final output
    }
}

// ---------------------------------------------------------------------------
extern "C" void kernel_launch(void* const* d_in, const int* in_sizes, int n_in,
                              void* d_out, int out_size, void* d_ws, size_t ws_size,
                              hipStream_t stream) {
    (void)in_sizes; (void)n_in; (void)out_size; (void)ws_size;
    const float* q  = (const float*)d_in[0];
    const float* k  = (const float*)d_in[1];
    const float* Wq = (const float*)d_in[2];
    const float* bq = (const float*)d_in[3];
    const float* Wk = (const float*)d_in[4];
    const float* bk = (const float*)d_in[5];
    float* out = (float*)d_out;

    const size_t nX = (size_t)kB * kL * kH;   // 8.4M elements (q or k)
    const size_t nW = (size_t)kH * kH;        // 1M elements (weights)

    // workspace layout (~143 MB): qs,ks f32 then bf16 hi/lo planes
    float* qs = (float*)d_ws;
    float* ks = qs + nX;
    unsigned short* p = (unsigned short*)(ks + nX);
    unsigned short* qhi  = p;            unsigned short* qlo  = qhi  + nX;
    unsigned short* khi  = qlo  + nX;    unsigned short* klo  = khi  + nX;
    unsigned short* Wqhi = klo  + nX;    unsigned short* Wqlo = Wqhi + nW;
    unsigned short* Wkhi = Wqlo + nW;    unsigned short* Wklo = Wkhi + nW;

    const int gX = (int)(nX / 4 + 255) / 256;
    const int gW = (int)(nW / 4 + 255) / 256;
    split_convert_kernel<<<gX, 256, 0, stream>>>(q,  qhi,  qlo,  (int)(nX / 4));
    split_convert_kernel<<<gX, 256, 0, stream>>>(k,  khi,  klo,  (int)(nX / 4));
    split_convert_kernel<<<gW, 256, 0, stream>>>(Wq, Wqhi, Wqlo, (int)(nW / 4));
    split_convert_kernel<<<gW, 256, 0, stream>>>(Wk, Wkhi, Wklo, (int)(nW / 4));

    dim3 g1((kB * kL) / BM, kH / BN, 2);     // 128 x 8 x 2 blocks
    proj_gemm_kernel<<<g1, dim3(256), 0, stream>>>(qhi, qlo, khi, klo,
                                                   Wqhi, Wqlo, Wkhi, Wklo,
                                                   bq, bk, qs, ks);

    band_softmax_kernel<<<dim3(kB * kL), dim3(288), 0, stream>>>(qs, ks, out);
}
